// BinaryBasicBlock_59210419143121
// MI455X (gfx1250) — 8-bit matrix operand order repaired and checked by reading, compiled, not run on hardware
//
#include <hip/hip_runtime.h>
#include <hip/hip_bf16.h>

typedef int v8i __attribute__((ext_vector_type(8)));

#define IN_CHN 64
#define OUT_CHN 64
#define HH 112
#define WW 112
#define NBATCH 32
#define KDIM 576
#define KSTEPS 9
#define MTILE 16
#define BN_EPS 1e-5f

#define OFF_BPACK 0
#define OFF_SCALE 36864
#define OFF_SUM   37120
#define OFF_SQ    37376

__global__ void prep_kernel(const float* __restrict__ wflat, float* __restrict__ wsf) {
    int t = threadIdx.x;
    if (t < OUT_CHN) {
        float s = 0.f;
        for (int i = 0; i < KDIM; ++i) s += fabsf(wflat[t * KDIM + i]);
        wsf[OFF_SCALE / 4 + t] = s * (1.0f / (float)KDIM);
        wsf[OFF_SUM   / 4 + t] = 0.f;
        wsf[OFF_SQ    / 4 + t] = 0.f;
    }
}

__global__ void packb_kernel(const float* __restrict__ wflat, int* __restrict__ bpack) {
    const int total = KSTEPS * 4 * 32 * 8;
    for (int dw = blockIdx.x * blockDim.x + threadIdx.x; dw < total;
         dw += gridDim.x * blockDim.x) {
        int kb   = dw / (4 * 32 * 8);
        int r    = dw % (4 * 32 * 8);
        int nt   = r / (32 * 8);
        int r2   = r % (32 * 8);
        int lane = r2 / 8;
        int v    = r2 % 8;
        int o     = nt * 16 + (lane & 15);
        int khalf = lane >> 4;
        unsigned w32 = 0;
        for (int j = 0; j < 4; ++j) {
            int c = khalf * 8 + (v >> 1) * 16 + (v & 1) * 4 + j;
            float wv = wflat[o * KDIM + c * 9 + kb];
            int s = (wv > 0.f) - (wv < 0.f);
            w32 |= ((unsigned)(s & 0xff)) << (8 * j);
        }
        bpack[dw] = (int)w32;
    }
}

__global__ __launch_bounds__(128) void conv_wmma_kernel(
        const float* __restrict__ x, const int* __restrict__ bpack,
        const float* __restrict__ scale, float* __restrict__ y,
        float* __restrict__ gsum, float* __restrict__ gsq) {
    __shared__ __align__(16) signed char halo[3 * 18 * IN_CHN];
    __shared__ float lds_sum[OUT_CHN];
    __shared__ float lds_sq[OUT_CHN];

    const int tid  = threadIdx.x;
    const int lane = tid & 31;
    const int nt   = tid >> 5;
    const int w0   = blockIdx.x * MTILE;
    const int h    = blockIdx.y;
    const int n    = blockIdx.z;

    for (int i = tid; i < IN_CHN * 3 * 18; i += 128) {
        int c   = i / 54, pos = i % 54;
        int hh  = pos / 18, ww = pos % 18;
        int gh  = h + hh - 1, gw = w0 + ww - 1;
        int s = 0;
        if (gh >= 0 && gh < HH && gw >= 0 && gw < WW) {
            float v = x[((n * IN_CHN + c) * HH + gh) * WW + gw];
            s = (v > 0.f) - (v < 0.f);
        }
        halo[(hh * 18 + ww) * 64 + c] = (signed char)s;
    }
    __syncthreads();

    const int m   = lane & 15;
    const int hiK = (lane >> 4) * 8;

    v8i acc = {0, 0, 0, 0, 0, 0, 0, 0};
#pragma unroll
    for (int kb = 0; kb < KSTEPS; ++kb) {
        const int kh = kb / 3;
        const int kw = kb - kh * 3;
        const int* p = (const int*)&halo[((kh * 18) + (m + kw)) * 64 + hiK];
        v8i a;
        a[0] = p[0];  a[1] = p[1];
        a[2] = p[4];  a[3] = p[5];
        a[4] = p[8];  a[5] = p[9];
        a[6] = p[12]; a[7] = p[13];

        const v8i* bp = (const v8i*)(bpack + ((kb * 4 + nt) * 32 + lane) * 8);
        v8i bfrag = *bp;

        acc = __builtin_amdgcn_wmma_i32_16x16x64_iu8(true, a, true, bfrag, acc,
                                                     false, false);
    }

    const int o    = nt * 16 + (lane & 15);
    const float sc = scale[o];
    float s1 = 0.f, s2 = 0.f;
#pragma unroll
    for (int r = 0; r < 8; ++r) {
        int   mr = r + ((lane >> 4) << 3);
        float yv = (float)acc[r] * sc;
        y[((n * OUT_CHN + o) * HH + h) * WW + (w0 + mr)] = yv;
        s1 += yv;
        s2 += yv * yv;
    }
    s1 += __shfl_down(s1, 16);
    s2 += __shfl_down(s2, 16);
    if (lane < 16) { lds_sum[o] = s1; lds_sq[o] = s2; }
    __syncthreads();
    if (tid < OUT_CHN) {
        atomicAdd(&gsum[tid], lds_sum[tid]);
        atomicAdd(&gsq[tid],  lds_sq[tid]);
    }
}

__global__ void bn_res_kernel(const float* __restrict__ x,
                              const float* __restrict__ gamma,
                              const float* __restrict__ beta,
                              const float* __restrict__ gsum,
                              const float* __restrict__ gsq,
                              float* __restrict__ out, int total) {
    int i = blockIdx.x * blockDim.x + threadIdx.x;
    if (i >= total) return;
    int o = (i / (HH * WW)) % OUT_CHN;
    const float inv_cnt = 1.0f / (float)(NBATCH * HH * WW);
    float mean = gsum[o] * inv_cnt;
    float var  = gsq[o] * inv_cnt - mean * mean;
    float inv  = gamma[o] * rsqrtf(var + BN_EPS);
    float yv   = out[i];
    out[i] = (yv - mean) * inv + beta[o] + x[i];
}

extern "C" void kernel_launch(void* const* d_in, const int* in_sizes, int n_in,
                              void* d_out, int out_size, void* d_ws, size_t ws_size,
                              hipStream_t stream) {
    const float* x     = (const float*)d_in[0];
    const float* wflat = (const float*)d_in[1];
    const float* gamma = (const float*)d_in[2];
    const float* beta  = (const float*)d_in[3];
    float* out = (float*)d_out;

    float* wsf   = (float*)d_ws;
    int*   bpack = (int*)((char*)d_ws + OFF_BPACK);
    float* scale = wsf + OFF_SCALE / 4;
    float* gsum  = wsf + OFF_SUM / 4;
    float* gsq   = wsf + OFF_SQ / 4;

    hipLaunchKernelGGL(prep_kernel, dim3(1), dim3(128), 0, stream, wflat, wsf);
    hipLaunchKernelGGL(packb_kernel, dim3(36), dim3(256), 0, stream, wflat, bpack);

    dim3 cgrid(WW / MTILE, HH, NBATCH);
    hipLaunchKernelGGL(conv_wmma_kernel, cgrid, dim3(128), 0, stream,
                       x, bpack, scale, out, gsum, gsq);

    int total = NBATCH * OUT_CHN * HH * WW;
    hipLaunchKernelGGL(bn_res_kernel, dim3((total + 255) / 256), dim3(256), 0,
                       stream, x, gamma, beta, gsum, gsq, out, total);
}
